// OneToManyRNN_64312840291068
// MI455X (gfx1250) — matrix operand order repaired and checked by reading, not compiled, not run on hardware
//
#include <hip/hip_runtime.h>
#include <hip/hip_bf16.h>
#include <math.h>

// ---- problem constants (match reference setup_inputs) ----
#define B_  64
#define T_  2048
#define I_  64
#define H_  128
#define O_  64
#define H2_ 256

typedef __attribute__((ext_vector_type(16))) __bf16 v16bf;
typedef __attribute__((ext_vector_type(8)))  __bf16 v8bf;
typedef __attribute__((ext_vector_type(8)))  float  v8f;

// D = A(16x32 bf16) * B(32x16 bf16) + C(16x16 f32)
static __device__ inline v8f wmma_bf16(v16bf a, v16bf b, v8f c) {
  return __builtin_amdgcn_wmma_f32_16x16x32_bf16(
      /*neg_a=*/false, a, /*neg_b=*/false, b,
      /*c_mod=*/(short)0, c, /*reuse_a=*/false, /*reuse_b=*/false);
}

// Hardware tanh (V_TANH_F32 on gfx1250) with safe fallback.
static __device__ inline float fast_tanh(float x) {
#if __has_builtin(__builtin_amdgcn_tanhf)
  return __builtin_amdgcn_tanhf(x);
#elif __has_builtin(__builtin_amdgcn_tanh_f32)
  return __builtin_amdgcn_tanh_f32(x);
#else
  return tanhf(x);
#endif
}

// Convert 16 contiguous f32 (16B-aligned) -> v16bf
static __device__ inline v16bf cvt16(const float* __restrict__ p) {
  v16bf o;
#pragma unroll
  for (int j = 0; j < 4; ++j) {
    float4 v = ((const float4*)p)[j];
    o[4 * j + 0] = (__bf16)v.x;
    o[4 * j + 1] = (__bf16)v.y;
    o[4 * j + 2] = (__bf16)v.z;
    o[4 * j + 3] = (__bf16)v.w;
  }
  return o;
}

// A-fragment (16x32 bf16) from f32 row-major source, row = m0+(lane&15).
// Layout: lanes 0-15: e0..15 -> K=kb+0..15
//         lanes16-31: e0..15 -> K=kb+16..31
static __device__ inline v16bf load_a_f32(const float* __restrict__ row,
                                          int kbase, int lane) {
  const int off = kbase + ((lane & 16) ? 16 : 0);
  v16bf a;
#pragma unroll
  for (int j = 0; j < 2; ++j) {
    float4 v = ((const float4*)(row + off))[j];
    a[4 * j + 0] = (__bf16)v.x; a[4 * j + 1] = (__bf16)v.y;
    a[4 * j + 2] = (__bf16)v.z; a[4 * j + 3] = (__bf16)v.w;
  }
#pragma unroll
  for (int j = 0; j < 2; ++j) {
    float4 v = ((const float4*)(row + off + 8))[j];
    a[8 + 4 * j + 0] = (__bf16)v.x; a[8 + 4 * j + 1] = (__bf16)v.y;
    a[8 + 4 * j + 2] = (__bf16)v.z; a[8 + 4 * j + 3] = (__bf16)v.w;
  }
  return a;
}

// A-fragment from a bf16 [.][ldm] matrix (LDS or global): two 16B loads
static __device__ inline v16bf load_a_bf16(const __bf16* __restrict__ h, int ldm,
                                           int m0, int kbase, int lane) {
  const int row = m0 + (lane & 15);
  const int off = kbase + ((lane & 16) ? 16 : 0);
  v8bf lo = *(const v8bf*)(h + row * ldm + off);
  v8bf hi = *(const v8bf*)(h + row * ldm + off + 8);
  v16bf a;
#pragma unroll
  for (int e = 0; e < 8; ++e) { a[e] = lo[e]; a[e + 8] = hi[e]; }
  return a;
}

// B-fragment (32x16 bf16) from a [N][K]-row-major bf16 weight (B[k][n] = W[n][k]).
static __device__ inline v16bf load_b_lds(const __bf16* __restrict__ w, int K,
                                          int n0, int kbase, int lane) {
  const int n  = n0 + (lane & 15);
  const int k0 = kbase + ((lane & 16) ? 16 : 0);
  return *(const v16bf*)(w + n * K + k0);
}

// Async global->LDS: each lane moves 32B (two b128), tracked by ASYNCcnt.
static __device__ inline void async_copy32(unsigned lds_byte_addr,
                                           const float* __restrict__ g) {
  asm volatile(
      "global_load_async_to_lds_b128 %0, %1, off\n\t"
      "global_load_async_to_lds_b128 %0, %1, off offset:16"
      :
      : "v"(lds_byte_addr), "v"((unsigned long long)(uintptr_t)g)
      : "memory");
}
static __device__ inline void wait_async0() {
  asm volatile("s_wait_asynccnt 0" ::: "memory");
}

// ------------------------------------------------------------------
// Kernel 1: xp[d][t*B+b][h] = x[b][t][:] @ W_ih[d]^T + b_ih[d]
// grid (1024, 2), block 256. One 16(M)x128(N) strip per wave, K=64.
// ------------------------------------------------------------------
__global__ void __launch_bounds__(256) birnn_xproj(
    const float* __restrict__ x,
    const float* __restrict__ Wih_f, const float* __restrict__ bih_f,
    const float* __restrict__ Wih_b, const float* __restrict__ bih_b,
    float* __restrict__ xp_f, float* __restrict__ xp_b) {
  const int dir = blockIdx.y;
  const float* W    = dir ? Wih_b : Wih_f;
  const float* bias = dir ? bih_b : bih_f;
  float*       xp   = dir ? xp_b  : xp_f;

  __shared__ __align__(32) __bf16 wl[H_ * I_];  // 16 KB
  for (int i = threadIdx.x; i < H_ * I_; i += 256) wl[i] = (__bf16)W[i];
  __syncthreads();

  const int lane = threadIdx.x & 31;
  const int wave = threadIdx.x >> 5;
  const int m0   = (blockIdx.x * 8 + wave) * 16;  // row in (t*B+b) space
  const int col  = lane & 15;

  const int mrowA = m0 + (lane & 15);
  const int tA = mrowA / B_, bA = mrowA % B_;
  const float* xrow = x + ((size_t)bA * T_ + tA) * I_;

  v8f c[8];
#pragma unroll
  for (int nt = 0; nt < 8; ++nt) {
    const float bv = bias[nt * 16 + col];
#pragma unroll
    for (int r = 0; r < 8; ++r) c[nt][r] = bv;
  }

#pragma unroll
  for (int kt = 0; kt < 2; ++kt) {
    v16bf a = load_a_f32(xrow, kt * 32, lane);
#pragma unroll
    for (int nt = 0; nt < 8; ++nt) {
      v16bf b = load_b_lds(wl, I_, nt * 16, kt * 32, lane);
      c[nt] = wmma_bf16(a, b, c[nt]);
    }
  }

  const int rbase = (lane & 16) ? 8 : 0;
#pragma unroll
  for (int nt = 0; nt < 8; ++nt)
#pragma unroll
    for (int r = 0; r < 8; ++r)
      xp[(size_t)(m0 + rbase + r) * H_ + nt * 16 + col] = c[nt][r];
}

// ------------------------------------------------------------------
// Kernel 2: sequential scan. grid (2) [dir], block 1024 = 32 waves.
// Wave (mt,nt) owns a 16x16 tile of h (64x128). W_hh^T frags live in
// registers, h ping-pongs as bf16 in LDS, and the next timestep's xp
// tile is prefetched into a double-buffered LDS stage with
// global_load_async_to_lds_b128 (ASYNCcnt) so the fetch latency hides
// under the WMMA+tanh of the current step. One barrier per timestep.
// ------------------------------------------------------------------
__global__ void __launch_bounds__(1024) birnn_scan(
    const float* __restrict__ xp_f, const float* __restrict__ xp_b,
    const float* __restrict__ Whh_f, const float* __restrict__ bhh_f,
    const float* __restrict__ Whh_b, const float* __restrict__ bhh_b,
    __bf16* __restrict__ y) {
  const int dir = blockIdx.x;
  const float* xp  = dir ? xp_b  : xp_f;
  const float* Whh = dir ? Whh_b : Whh_f;
  const float* bhh = dir ? bhh_b : bhh_f;

  __shared__ __align__(32) float  xstage[2][B_ * H_];  // 2 x 32 KB async stage
  __shared__ __align__(32) __bf16 hbuf[2][B_ * H_];    // 2 x 16 KB state

  const int tid  = threadIdx.x;
  const int lane = tid & 31;
  const int wave = tid >> 5;          // 0..31
  const int m0 = (wave >> 3) * 16;    // batch-row tile
  const int n0 = (wave & 7) * 16;     // H-column tile
  const int col = lane & 15;
  const int rbase = (lane & 16) ? 8 : 0;

  // LDS byte addresses of the two stage buffers for this thread's 32B slice
  const unsigned st_addr[2] = {
      (unsigned)(uintptr_t)(const void*)&xstage[0][0] + (unsigned)tid * 32u,
      (unsigned)(uintptr_t)(const void*)&xstage[1][0] + (unsigned)tid * 32u};

  // Register-resident W_hh^T B-fragments (4 k-steps of 32)
  v16bf Wf[4];
#pragma unroll
  for (int kt = 0; kt < 4; ++kt) {
    const int n  = n0 + col;
    const int k0 = kt * 32 + ((lane & 16) ? 16 : 0);
    Wf[kt] = cvt16(Whh + n * H_ + k0);
  }
  const float bv = bhh[n0 + col];

  // Prefetch xp tile for the first timestep into stage 0.
  {
    const int t0 = dir ? (T_ - 1) : 0;
    async_copy32(st_addr[0], xp + (size_t)t0 * B_ * H_ + tid * 8);
  }
  for (int i = tid; i < B_ * H_; i += 1024) {
    hbuf[0][i] = (__bf16)0.f;
    hbuf[1][i] = (__bf16)0.f;
  }
  wait_async0();
  __syncthreads();

  int p = 0, q = 0;
  for (int s = 0; s < T_; ++s) {
    const int t = dir ? (T_ - 1 - s) : s;

    // C = xp[t] tile (from LDS stage) + b_hh
    const float* sx = &xstage[q][0];
    v8f c;
#pragma unroll
    for (int r = 0; r < 8; ++r)
      c[r] = sx[(m0 + rbase + r) * H_ + n0 + col] + bv;

    // Kick off async prefetch of the next timestep's xp tile.
    if (s + 1 < T_) {
      const int tn = dir ? (T_ - 2 - s) : (s + 1);
      async_copy32(st_addr[q ^ 1], xp + (size_t)tn * B_ * H_ + tid * 8);
    }

    // C += h @ W_hh^T  (K = 128 -> 4 WMMAs on the dependent chain)
#pragma unroll
    for (int kt = 0; kt < 4; ++kt) {
      v16bf a = load_a_bf16(&hbuf[p][0], H_, m0, kt * 32, lane);
      c = wmma_bf16(a, Wf[kt], c);
    }

    // h = tanh(C); emit y (bf16, time-aligned) and next-state tile
    __bf16* yrow = y + (size_t)t * B_ * H2_ + dir * H_;
#pragma unroll
    for (int r = 0; r < 8; ++r) {
      const float hv = fast_tanh(c[r]);
      const int m = m0 + rbase + r;
      const __bf16 hb = (__bf16)hv;
      yrow[(size_t)m * H2_ + n0 + col] = hb;
      hbuf[p ^ 1][m * H_ + n0 + col] = hb;
    }

    wait_async0();     // own prefetch done (covered by wmma/tanh window)
    __syncthreads();   // everyone's stage + h writes visible
    p ^= 1;
    q ^= 1;
  }
}

// ------------------------------------------------------------------
// Kernel 3: out[b][t][o] = y[t*B+b][:] @ W_out^T + b_out   (y is bf16)
// grid (1024), block 256. One 16(M)x64(N) strip per wave, K=256.
// ------------------------------------------------------------------
__global__ void __launch_bounds__(256) birnn_out(
    const __bf16* __restrict__ y, const float* __restrict__ Wout,
    const float* __restrict__ bout, float* __restrict__ out) {
  __shared__ __align__(32) __bf16 wl[O_ * H2_];  // 32 KB
  for (int i = threadIdx.x; i < O_ * H2_; i += 256) wl[i] = (__bf16)Wout[i];
  __syncthreads();

  const int lane = threadIdx.x & 31;
  const int wave = threadIdx.x >> 5;
  const int m0   = (blockIdx.x * 8 + wave) * 16;
  const int col  = lane & 15;

  v8f c[4];
#pragma unroll
  for (int nt = 0; nt < 4; ++nt) {
    const float bv = bout[nt * 16 + col];
#pragma unroll
    for (int r = 0; r < 8; ++r) c[nt][r] = bv;
  }

#pragma unroll
  for (int kt = 0; kt < 8; ++kt) {
    v16bf a = load_a_bf16(y, H2_, m0, kt * 32, lane);
#pragma unroll
    for (int nt = 0; nt < 4; ++nt) {
      v16bf b = load_b_lds(wl, H2_, nt * 16, kt * 32, lane);
      c[nt] = wmma_bf16(a, b, c[nt]);
    }
  }

  const int rbase = (lane & 16) ? 8 : 0;
#pragma unroll
  for (int nt = 0; nt < 4; ++nt)
#pragma unroll
    for (int r = 0; r < 8; ++r) {
      const int m = m0 + rbase + r;   // m = t*B + b
      const int t = m / B_, b = m % B_;
      out[((size_t)b * T_ + t) * O_ + nt * 16 + col] = c[nt][r];
    }
}

// ------------------------------------------------------------------
extern "C" void kernel_launch(void* const* d_in, const int* in_sizes, int n_in,
                              void* d_out, int out_size, void* d_ws, size_t ws_size,
                              hipStream_t stream) {
  const float* x     = (const float*)d_in[0];
  const float* Wih_f = (const float*)d_in[1];
  const float* Whh_f = (const float*)d_in[2];
  const float* bih_f = (const float*)d_in[3];
  const float* bhh_f = (const float*)d_in[4];
  const float* Wih_b = (const float*)d_in[5];
  const float* Whh_b = (const float*)d_in[6];
  const float* bih_b = (const float*)d_in[7];
  const float* bhh_b = (const float*)d_in[8];
  const float* Wout  = (const float*)d_in[9];
  const float* bout  = (const float*)d_in[10];
  float* out = (float*)d_out;

  // workspace: xp_f (64MB f32) | xp_b (64MB f32) | y (67MB bf16)
  float*  xp_f = (float*)d_ws;
  float*  xp_b = xp_f + (size_t)T_ * B_ * H_;
  __bf16* yy   = (__bf16*)(xp_b + (size_t)T_ * B_ * H_);

  birnn_xproj<<<dim3(1024, 2), dim3(256), 0, stream>>>(
      x, Wih_f, bih_f, Wih_b, bih_b, xp_f, xp_b);
  birnn_scan<<<dim3(2), dim3(1024), 0, stream>>>(
      xp_f, xp_b, Whh_f, bhh_f, Whh_b, bhh_b, yy);
  birnn_out<<<dim3(1024), dim3(256), 0, stream>>>(yy, Wout, bout, out);
}
